// PtHeadSelection_22531398435396
// MI455X (gfx1250) — matrix operand order repaired and checked by reading, not compiled, not run on hardware
//
#include <hip/hip_runtime.h>

#define DEV __device__ __forceinline__

typedef __bf16 bf16_t;
typedef __attribute__((ext_vector_type(16))) __bf16 v16bf;
typedef __attribute__((ext_vector_type(8)))  float  v8f;

constexpr int Bc = 2;
constexpr int Sc = 2048;
constexpr int Dc = 768;
constexpr int Cc = 8;
constexpr int Rc = 96;

// ---------- feature detection: Tensor Data Mover ----------
#ifdef __has_builtin
#if __has_builtin(__builtin_amdgcn_tensor_load_to_lds) && \
    __has_builtin(__builtin_amdgcn_s_wait_tensorcnt)
#define USE_TDM 1
#endif
#endif
#ifndef USE_TDM
#define USE_TDM 0
#endif

// ---------- bf16 <-> f32 helpers (bit-level) ----------
DEV float bf2f(bf16_t h) {
  unsigned short s = __builtin_bit_cast(unsigned short, h);
  unsigned int u = ((unsigned int)s) << 16;
  return __builtin_bit_cast(float, u);
}
DEV bf16_t f2bf(float x) {
  unsigned int u = __builtin_bit_cast(unsigned int, x);
  unsigned int r = u + 0x7FFFu + ((u >> 16) & 1u);   // round-nearest-even
  unsigned short s = (unsigned short)(r >> 16);
  return __builtin_bit_cast(bf16_t, s);
}

// ---------- WMMA fragment loaders (CDNA5 wave32 layouts, 16x16x32 bf16) ----------
// A fragment: 16(M) x 32(K) row-major at p, row stride ld.
DEV v16bf frag_a_rm(const bf16_t* p, int ld) {
  int lane = threadIdx.x & 31;
  int m = lane & 15, kb = (lane >> 4) << 3;
  const bf16_t* r0 = p + (size_t)m * ld + kb;
  v16bf f;
#pragma unroll
  for (int i = 0; i < 8; i++) { f[i] = r0[i]; f[8 + i] = r0[16 + i]; }
  return f;
}
// B fragment: 32(K) x 16(N), stored N-major: elem(k,n) = p[n*ld + k].
DEV v16bf frag_b_nk(const bf16_t* p, int ld) {
  int lane = threadIdx.x & 31;
  int n = lane & 15, kb = (lane >> 4) << 3;
  const bf16_t* r = p + (size_t)n * ld + kb;
  v16bf f;
#pragma unroll
  for (int i = 0; i < 8; i++) { f[i] = r[i]; f[8 + i] = r[16 + i]; }
  return f;
}
// A fragment from a 16x32 row-major LDS tile (row stride 32)
DEV v16bf frag_a_lds32(const bf16_t* As) {
  int lane = threadIdx.x & 31;
  int m = lane & 15, kb = (lane >> 4) << 3;
  const bf16_t* r = As + m * 32 + kb;
  v16bf f;
#pragma unroll
  for (int i = 0; i < 8; i++) { f[i] = r[i]; f[8 + i] = r[16 + i]; }
  return f;
}
DEV v8f wmma_bf(v16bf a, v16bf b, v8f c) {
  return __builtin_amdgcn_wmma_f32_16x16x32_bf16(false, a, false, b, (short)0, c,
                                                 false, false);
}

#if USE_TDM
// ---------- Tensor Data Mover: async 2D tile load global->LDS ----------
typedef __attribute__((ext_vector_type(4))) unsigned int u32x4;
typedef __attribute__((ext_vector_type(8))) int i32x8;
typedef __attribute__((ext_vector_type(4))) int i32x4;

DEV unsigned lds_off_of(const void* p) {
  // flat LDS addresses carry the LDS byte offset in the low 32 bits (ISA 10.2)
  return (unsigned)(unsigned long long)p;
}
// 2D tile: tile0 = contiguous extent (elements), tile1 = rows; 2-byte elements.
DEV void tdm_load_2d(unsigned lds_byte_off, const void* gptr,
                     unsigned tile0, unsigned tile1,
                     unsigned dim0, unsigned dim1, unsigned stride0) {
  unsigned long long ga = (unsigned long long)gptr;
  u32x4 g0;
  g0[0] = 1u;                                               // count=1 (valid user D#)
  g0[1] = lds_byte_off;                                     // lds_addr
  g0[2] = (unsigned)ga;                                     // global_addr[31:0]
  g0[3] = (unsigned)((ga >> 32) & 0x01FFFFFFu) | 0x80000000u; // [56:32] + type=2
  i32x8 g1;
  g1[0] = (int)0x00010000u;                                 // data_size=1 (2 bytes)
  g1[1] = (int)((dim0 & 0xFFFFu) << 16);                    // tensor_dim0[15:0]
  g1[2] = (int)(((dim0 >> 16) & 0xFFFFu) | ((dim1 & 0xFFFFu) << 16));
  g1[3] = (int)(((dim1 >> 16) & 0xFFFFu) | (tile0 << 16));  // tile_dim0
  g1[4] = (int)tile1;                                       // tile_dim1 (tile_dim2=0)
  g1[5] = (int)stride0;                                     // tensor_dim0_stride[31:0]
  g1[6] = 0;
  g1[7] = 0;
  i32x4 z4 = {0, 0, 0, 0};
#if defined(__clang_major__) && (__clang_major__ >= 23)
  i32x8 z8 = {0, 0, 0, 0, 0, 0, 0, 0};
  __builtin_amdgcn_tensor_load_to_lds(g0, g1, z4, z4, z8, 0);
#else
  __builtin_amdgcn_tensor_load_to_lds(g0, g1, z4, z4, 0);
#endif
}
#endif  // USE_TDM

// ---------- Kernel 0a: fp32 -> bf16 convert ----------
__global__ void k_conv(const float* __restrict__ in, bf16_t* __restrict__ out, int n) {
  int i = blockIdx.x * 256 + threadIdx.x;
  if (i < n) out[i] = f2bf(in[i]);
}

// ---------- Kernel 0b: U,V -> bf16 (plain + transposed copies) ----------
__global__ void k_convw(const float* __restrict__ U, const float* __restrict__ V,
                        bf16_t* __restrict__ Ub, bf16_t* __restrict__ Vb,
                        bf16_t* __restrict__ UbT, bf16_t* __restrict__ VbT) {
  int idx = blockIdx.x * 256 + threadIdx.x;
  if (idx >= Dc * Dc) return;
  int i = idx / Dc, j = idx % Dc;
  float u = U[idx], v = V[idx];
  Ub[idx] = f2bf(u);
  Vb[idx] = f2bf(v);
  UbT[(size_t)j * Dc + i] = f2bf(u);
  VbT[(size_t)j * Dc + i] = f2bf(v);
}

// ---------- Kernel 1: y_u = qz @ U^T, y_v = qz @ V^T (head-major bf16 out) ----------
__global__ void k_proj(const bf16_t* __restrict__ qz, const bf16_t* __restrict__ Ub,
                       const bf16_t* __restrict__ Vb, bf16_t* __restrict__ yu,
                       bf16_t* __restrict__ yv) {
  int m0 = blockIdx.x * 16;
  int n0 = blockIdx.y * 32;
  v8f z = {0, 0, 0, 0, 0, 0, 0, 0};
  v8f aU[2] = {z, z}, aV[2] = {z, z};
  for (int k0 = 0; k0 < Dc; k0 += 32) {
    v16bf a = frag_a_rm(qz + (size_t)m0 * Dc + k0, Dc);
    v16bf bu[2], bv[2];
#pragma unroll
    for (int j = 0; j < 2; j++) {
      bu[j] = frag_b_nk(Ub + (size_t)(n0 + 16 * j) * Dc + k0, Dc);
      bv[j] = frag_b_nk(Vb + (size_t)(n0 + 16 * j) * Dc + k0, Dc);
    }
#pragma unroll
    for (int j = 0; j < 2; j++) {
      aU[j] = wmma_bf(a, bu[j], aU[j]);
      aV[j] = wmma_bf(a, bv[j], aV[j]);
    }
  }
  int lane = threadIdx.x & 31, n = lane & 15, lh = lane >> 4;
#pragma unroll
  for (int j = 0; j < 2; j++)
#pragma unroll
    for (int i = 0; i < 8; i++) {
      int d = n0 + 16 * j + n;
      int row = m0 + lh * 8 + i;
      int b = row >> 11, s = row & (Sc - 1);
      int c = d / Rc, r = d % Rc;
      size_t o = (((size_t)b * Cc + c) * Sc + s) * Rc + r;
      yu[o] = f2bf(aU[j][i]);
      yv[o] = f2bf(aV[j][i]);
    }
}

// ---------- Kernel 2: elementwise RoPE -> ur (s,r), vr (s,r), vrT (r,s), uoT (r,s) ----------
__global__ void k_rope(const bf16_t* __restrict__ yu, const bf16_t* __restrict__ yv,
                       const float* __restrict__ cosp, const float* __restrict__ sinp,
                       bf16_t* __restrict__ ur, bf16_t* __restrict__ vr,
                       bf16_t* __restrict__ vrT, bf16_t* __restrict__ uoT) {
  size_t idx = (size_t)blockIdx.x * 256 + threadIdx.x;
  if (idx >= (size_t)Bc * Cc * Sc * Rc) return;
  int r = (int)(idx % Rc);
  size_t t = idx / Rc;
  int s = (int)(t % Sc);
  int h = (int)(t / Sc);
  int b = h >> 3;
  size_t ci = ((size_t)b * Sc + s) * Rc + r;
  float cs = cosp[ci], sn = sinp[ci];
  float xu = bf2f(yu[idx]), xv = bf2f(yv[idx]);
  size_t pidx = (r < 48) ? idx + 48 : idx - 48;
  float rotu = (r < 48) ? -bf2f(yu[pidx]) : bf2f(yu[pidx]);
  float rotv = (r < 48) ? -bf2f(yv[pidx]) : bf2f(yv[pidx]);
  float urv = xu * cs + rotu * sn;
  float uov = xu * cs - rotu * sn;
  float vrv = xv * cs + rotv * sn;
  ur[idx] = f2bf(urv);
  vr[idx] = f2bf(vrv);
  size_t tidx = ((size_t)h * Rc + r) * Sc + s;
  vrT[tidx] = f2bf(vrv);
  uoT[tidx] = f2bf(uov);
}

// ---------- Kernel 3: scores + online softmax (2-pass recompute) -> qh bf16 ----------
__global__ void k_scores(const bf16_t* __restrict__ ur, const bf16_t* __restrict__ vr,
                         bf16_t* __restrict__ qh) {
  int wg = blockIdx.x;
  int qt = wg & 127, head = wg >> 7;
  const bf16_t* urh = ur + (size_t)head * Sc * Rc;
  const bf16_t* vrh = vr + (size_t)head * Sc * Rc;
  bf16_t* qhh = qh + (size_t)head * Sc * Sc;
  int q0 = qt * 16;
  int wave = threadIdx.x >> 5, lane = threadIdx.x & 31, lh = lane >> 4;

  v16bf a0 = frag_a_rm(urh + (size_t)q0 * Rc + 0, Rc);
  v16bf a1 = frag_a_rm(urh + (size_t)q0 * Rc + 32, Rc);
  v16bf a2 = frag_a_rm(urh + (size_t)q0 * Rc + 64, Rc);

  __shared__ float wmax[4][16], wsum[4][16], gmax[16], gsum[16];

  float rmax[8], rsum[8];
#pragma unroll
  for (int i = 0; i < 8; i++) { rmax[i] = -3.0e38f; rsum[i] = 0.f; }

  int kt0 = wave * 32;
  for (int kt = kt0; kt < kt0 + 32; ++kt) {
    int k0 = kt * 16;
    const bf16_t* bp = vrh + (size_t)k0 * Rc;
    v8f acc = {0, 0, 0, 0, 0, 0, 0, 0};
    v16bf b0 = frag_b_nk(bp + 0, Rc), b1 = frag_b_nk(bp + 32, Rc),
          b2 = frag_b_nk(bp + 64, Rc);
    acc = wmma_bf(a0, b0, acc);
    acc = wmma_bf(a1, b1, acc);
    acc = wmma_bf(a2, b2, acc);
#pragma unroll
    for (int i = 0; i < 8; i++) {
      float v = acc[i] * 768.0f;
      float tmax = v;
      tmax = fmaxf(tmax, __shfl_xor(tmax, 1, 32));
      tmax = fmaxf(tmax, __shfl_xor(tmax, 2, 32));
      tmax = fmaxf(tmax, __shfl_xor(tmax, 4, 32));
      tmax = fmaxf(tmax, __shfl_xor(tmax, 8, 32));
      float nm = fmaxf(rmax[i], tmax);
      float e = __expf(v - nm);
      e += __shfl_xor(e, 1, 32);
      e += __shfl_xor(e, 2, 32);
      e += __shfl_xor(e, 4, 32);
      e += __shfl_xor(e, 8, 32);
      rsum[i] = rsum[i] * __expf(rmax[i] - nm) + e;
      rmax[i] = nm;
    }
  }
  if ((lane & 15) == 0) {
#pragma unroll
    for (int i = 0; i < 8; i++) { wmax[wave][lh * 8 + i] = rmax[i]; wsum[wave][lh * 8 + i] = rsum[i]; }
  }
  __syncthreads();
  if (threadIdx.x < 16) {
    int rr = threadIdx.x;
    float m = wmax[0][rr];
    for (int w = 1; w < 4; w++) m = fmaxf(m, wmax[w][rr]);
    float s = 0.f;
    for (int w = 0; w < 4; w++) s += wsum[w][rr] * __expf(wmax[w][rr] - m);
    gmax[rr] = m;
    gsum[rr] = 1.0f / s;
  }
  __syncthreads();
  float gm[8], gs[8];
#pragma unroll
  for (int i = 0; i < 8; i++) { gm[i] = gmax[lh * 8 + i]; gs[i] = gsum[lh * 8 + i]; }

  int n = lane & 15;
  for (int kt = kt0; kt < kt0 + 32; ++kt) {
    int k0 = kt * 16;
    const bf16_t* bp = vrh + (size_t)k0 * Rc;
    v8f acc = {0, 0, 0, 0, 0, 0, 0, 0};
    v16bf b0 = frag_b_nk(bp + 0, Rc), b1 = frag_b_nk(bp + 32, Rc),
          b2 = frag_b_nk(bp + 64, Rc);
    acc = wmma_bf(a0, b0, acc);
    acc = wmma_bf(a1, b1, acc);
    acc = wmma_bf(a2, b2, acc);
#pragma unroll
    for (int i = 0; i < 8; i++) {
      float p = __expf(acc[i] * 768.0f - gm[i]) * gs[i];
      qhh[(size_t)(q0 + lh * 8 + i) * Sc + k0 + n] = f2bf(p);
    }
  }
}

// ---------- Kernel 4: qh_v1 = rope_o(qh @ vr), M=32 per WG ----------
// qh tiles streamed via TDM (async tensor DMA) with LDS double buffering.
__global__ void k_attnv1(const bf16_t* __restrict__ qh, const bf16_t* __restrict__ vrT,
                         const float* __restrict__ cosp, const float* __restrict__ sinp,
                         bf16_t* __restrict__ hv1) {
  int wg = blockIdx.x;                       // B*C*(S/32) = 1024
  int qt = wg & 63, head = wg >> 6, b = head >> 3;
  const bf16_t* qhh = qh + (size_t)head * Sc * Sc;
  const bf16_t* vth = vrT + (size_t)head * Rc * Sc;
  bf16_t* outh = hv1 + (size_t)head * Sc * Rc;
  int q0 = qt * 32;
  v8f z = {0, 0, 0, 0, 0, 0, 0, 0};
  v8f acc[2][6];
#pragma unroll
  for (int t = 0; t < 2; t++)
#pragma unroll
    for (int j = 0; j < 6; j++) acc[t][j] = z;

  constexpr int KT = Sc / 32;                // 64 K-steps
#if USE_TDM
  __shared__ bf16_t Ab[2][32 * 32];
  tdm_load_2d(lds_off_of(Ab[0]), qhh + (size_t)q0 * Sc, 32, 32, Sc, Sc, Sc);
#endif
  for (int it = 0; it < KT; ++it) {
    int k0 = it * 32;
    v16bf a0, a1;
#if USE_TDM
    if (it + 1 < KT) {
      tdm_load_2d(lds_off_of(Ab[(it + 1) & 1]),
                  qhh + (size_t)q0 * Sc + (k0 + 32), 32, 32, Sc, Sc, Sc);
      __builtin_amdgcn_s_wait_tensorcnt(1);  // current buffer ready
    } else {
      __builtin_amdgcn_s_wait_tensorcnt(0);
    }
    const bf16_t* Ak = Ab[it & 1];
    a0 = frag_a_lds32(Ak);
    a1 = frag_a_lds32(Ak + 16 * 32);
#else
    __builtin_prefetch((const void*)(qhh + (size_t)q0 * Sc + k0 + 64), 0, 1);
    a0 = frag_a_rm(qhh + (size_t)q0 * Sc + k0, Sc);
    a1 = frag_a_rm(qhh + (size_t)(q0 + 16) * Sc + k0, Sc);
#endif
    v16bf bfr[6];
#pragma unroll
    for (int j = 0; j < 6; j++)
      bfr[j] = frag_b_nk(vth + (size_t)(j * 16) * Sc + k0, Sc);
#pragma unroll
    for (int j = 0; j < 6; j++) {
      acc[0][j] = wmma_bf(a0, bfr[j], acc[0][j]);
      acc[1][j] = wmma_bf(a1, bfr[j], acc[1][j]);
    }
  }
  int lane = threadIdx.x & 31, n = lane & 15, lh = lane >> 4;
#pragma unroll
  for (int t = 0; t < 2; t++)
#pragma unroll
    for (int j = 0; j < 6; j++) {
      int r = j * 16 + n;
      int pj = (j < 3) ? j + 3 : j - 3;
#pragma unroll
      for (int i = 0; i < 8; i++) {
        int srow = q0 + t * 16 + lh * 8 + i;
        size_t ci = ((size_t)b * Sc + srow) * Rc + r;
        float cs = cosp[ci], sn = sinp[ci];
        float x = acc[t][j][i], xp = acc[t][pj][i];
        float val = (r < 48) ? (x * cs + xp * sn) : (x * cs - xp * sn);  // rope_o
        outh[(size_t)srow * Rc + r] = f2bf(val);
      }
    }
}

// ---------- Kernel 5: qh_v2 = rope(qh^T @ uo), M=32 per WG ----------
// A = qh transposed per head: transposed through LDS each K step.
__global__ void k_attnv2(const bf16_t* __restrict__ qh, const bf16_t* __restrict__ uoT,
                         const float* __restrict__ cosp, const float* __restrict__ sinp,
                         bf16_t* __restrict__ hv2) {
  int wg = blockIdx.x;                       // 1024
  int qt = wg & 63, head = wg >> 6, b = head >> 3;
  const bf16_t* qhh = qh + (size_t)head * Sc * Sc;
  const bf16_t* uth = uoT + (size_t)head * Rc * Sc;
  bf16_t* outh = hv2 + (size_t)head * Sc * Rc;
  int q0 = qt * 32;
  int lane = threadIdx.x & 31;
  __shared__ bf16_t As[32 * 32];
  v8f z = {0, 0, 0, 0, 0, 0, 0, 0};
  v8f acc[2][6];
#pragma unroll
  for (int t = 0; t < 2; t++)
#pragma unroll
    for (int j = 0; j < 6; j++) acc[t][j] = z;

  for (int k0 = 0; k0 < Sc; k0 += 32) {
    // stage: As[m][kk] = qh[k0+kk][q0+m]  (lane kk copies one contiguous row chunk)
    const bf16_t* src = qhh + (size_t)(k0 + lane) * Sc + q0;
    __builtin_prefetch((const void*)(src + 32 * Sc), 0, 1);
#pragma unroll
    for (int m = 0; m < 32; m++) As[m * 32 + lane] = src[m];
    __syncthreads();
    v16bf a0 = frag_a_lds32(As);
    v16bf a1 = frag_a_lds32(As + 16 * 32);
    v16bf bfr[6];
#pragma unroll
    for (int j = 0; j < 6; j++)
      bfr[j] = frag_b_nk(uth + (size_t)(j * 16) * Sc + k0, Sc);
#pragma unroll
    for (int j = 0; j < 6; j++) {
      acc[0][j] = wmma_bf(a0, bfr[j], acc[0][j]);
      acc[1][j] = wmma_bf(a1, bfr[j], acc[1][j]);
    }
    __syncthreads();
  }
  int n = lane & 15, lh = lane >> 4;
#pragma unroll
  for (int t = 0; t < 2; t++)
#pragma unroll
    for (int j = 0; j < 6; j++) {
      int r = j * 16 + n;
      int pj = (j < 3) ? j + 3 : j - 3;
#pragma unroll
      for (int i = 0; i < 8; i++) {
        int srow = q0 + t * 16 + lh * 8 + i;
        size_t ci = ((size_t)b * Sc + srow) * Rc + r;
        float cs = cosp[ci], sn = sinp[ci];
        float x = acc[t][j][i], xp = acc[t][pj][i];
        float val = (r < 48) ? (x * cs - xp * sn) : (x * cs + xp * sn);  // rope
        outh[(size_t)srow * Rc + r] = f2bf(val);
      }
    }
}

// ---------- Kernel 6: out = qh_v1' @ U + qh_v2' @ V (fp32 out) ----------
__global__ void k_final(const bf16_t* __restrict__ hv1, const bf16_t* __restrict__ hv2,
                        const bf16_t* __restrict__ UbT, const bf16_t* __restrict__ VbT,
                        float* __restrict__ out) {
  int m0 = blockIdx.x * 16;
  int n0 = blockIdx.y * 32;
  int b = m0 >> 11, s0 = m0 & (Sc - 1);
  v8f z = {0, 0, 0, 0, 0, 0, 0, 0};
  v8f acc[2] = {z, z};
  for (int k0 = 0; k0 < Dc; k0 += 32) {
    int c = k0 / Rc, r0 = k0 % Rc;
    size_t abase = (((size_t)b * Cc + c) * Sc + s0) * Rc + r0;
    v16bf a1 = frag_a_rm(hv1 + abase, Rc);
    v16bf a2 = frag_a_rm(hv2 + abase, Rc);
    v16bf bu[2], bv[2];
#pragma unroll
    for (int j = 0; j < 2; j++) {
      bu[j] = frag_b_nk(UbT + (size_t)(n0 + 16 * j) * Dc + k0, Dc);
      bv[j] = frag_b_nk(VbT + (size_t)(n0 + 16 * j) * Dc + k0, Dc);
    }
#pragma unroll
    for (int j = 0; j < 2; j++) {
      acc[j] = wmma_bf(a1, bu[j], acc[j]);
      acc[j] = wmma_bf(a2, bv[j], acc[j]);
    }
  }
  int lane = threadIdx.x & 31, n = lane & 15, lh = lane >> 4;
#pragma unroll
  for (int j = 0; j < 2; j++)
#pragma unroll
    for (int i = 0; i < 8; i++)
      out[(size_t)(m0 + lh * 8 + i) * Dc + n0 + 16 * j + n] = acc[j][i];
}

// ---------- Host launcher ----------
extern "C" void kernel_launch(void* const* d_in, const int* in_sizes, int n_in,
                              void* d_out, int out_size, void* d_ws, size_t ws_size,
                              hipStream_t stream) {
  (void)in_sizes; (void)n_in; (void)out_size; (void)ws_size;
  const float* qz   = (const float*)d_in[0];
  const float* cosp = (const float*)d_in[1];
  const float* sinp = (const float*)d_in[2];
  const float* U    = (const float*)d_in[3];
  const float* V    = (const float*)d_in[4];
  float* out = (float*)d_out;

  char* w = (char*)d_ws;
  size_t off = 0;
  auto alloc = [&](size_t bytes) -> void* {
    void* p = w + off;
    off += (bytes + 255) & ~(size_t)255;
    return p;
  };
  const size_t NX = (size_t)Bc * Sc * Dc;
  const size_t NW = (size_t)Dc * Dc;
  const size_t NH = (size_t)Bc * Cc * Sc * Rc;
  const size_t NQ = (size_t)Bc * Cc * Sc * Sc;

  bf16_t* qzb = (bf16_t*)alloc(NX * 2);
  bf16_t* Ub  = (bf16_t*)alloc(NW * 2);
  bf16_t* Vb  = (bf16_t*)alloc(NW * 2);
  bf16_t* UbT = (bf16_t*)alloc(NW * 2);
  bf16_t* VbT = (bf16_t*)alloc(NW * 2);
  bf16_t* yu  = (bf16_t*)alloc(NH * 2);
  bf16_t* yv  = (bf16_t*)alloc(NH * 2);
  bf16_t* ur  = (bf16_t*)alloc(NH * 2);
  bf16_t* vr  = (bf16_t*)alloc(NH * 2);
  bf16_t* vrT = (bf16_t*)alloc(NH * 2);
  bf16_t* uoT = (bf16_t*)alloc(NH * 2);
  bf16_t* qh  = (bf16_t*)alloc(NQ * 2);
  bf16_t* hv1 = (bf16_t*)alloc(NH * 2);
  bf16_t* hv2 = (bf16_t*)alloc(NH * 2);

  k_conv<<<(int)((NX + 255) / 256), 256, 0, stream>>>(qz, qzb, (int)NX);
  k_convw<<<(int)((NW + 255) / 256), 256, 0, stream>>>(U, V, Ub, Vb, UbT, VbT);
  k_proj<<<dim3((Bc * Sc) / 16, Dc / 32), 32, 0, stream>>>(qzb, Ub, Vb, yu, yv);
  k_rope<<<(int)((NH + 255) / 256), 256, 0, stream>>>(yu, yv, cosp, sinp, ur, vr, vrT, uoT);
  k_scores<<<Bc * Cc * (Sc / 16), 128, 0, stream>>>(ur, vr, qh);
  k_attnv1<<<Bc * Cc * (Sc / 32), 32, 0, stream>>>(qh, vrT, cosp, sinp, hv1);
  k_attnv2<<<Bc * Cc * (Sc / 32), 32, 0, stream>>>(qh, uoT, cosp, sinp, hv2);
  k_final<<<dim3((Bc * Sc) / 16, Dc / 32), 32, 0, stream>>>(hv1, hv2, UbT, VbT, out);
}
